// MSAColumnAttention_52501680226865
// MI455X (gfx1250) — matrix operand order repaired and checked by reading, not compiled, not run on hardware
//
#include <hip/hip_runtime.h>
#include <math.h>

typedef __attribute__((ext_vector_type(16))) __bf16 v16bf;
typedef __attribute__((ext_vector_type(8)))  float  v8f;
typedef __attribute__((ext_vector_type(4)))  unsigned int v4u;
typedef __attribute__((ext_vector_type(8)))  int v8i;
typedef __attribute__((ext_vector_type(4)))  int v4i;
typedef __bf16 bf16_t;

#define S_DIM 256
#define I_DIM 256
#define C_DIM 256
#define H_DIM 8
#define D_DIM 32
#define NROW  (S_DIM * I_DIM)   // 65536
#define SP    257               // padded f32 row pitch for score matrix

// Generic pointer -> 32-bit LDS byte offset (for async/TDM destination operands).
__device__ inline unsigned to_lds(const void* p) {
  return (unsigned)(unsigned long long)(__attribute__((address_space(3))) const void*)p;
}

// Async global->LDS 128b copy (ASYNCcnt path, no VGPR staging).
__device__ inline void async_copy_b128(unsigned lds_off, const void* gptr) {
  unsigned long long ga = (unsigned long long)gptr;
  asm volatile("global_load_async_to_lds_b128 %0, %1, off"
               :: "v"(lds_off), "v"(ga) : "memory");
}
__device__ inline void async_wait0() {
  asm volatile("s_wait_asynccnt 0" ::: "memory");
}

// Tensor Data Mover: contiguous 16KB (8192 x 2B) slab, global -> LDS.
// D# packing per CDNA5 ISA.
__device__ inline void tdm_load_16kb(unsigned lds_off, const void* gptr) {
  unsigned long long ga = (unsigned long long)gptr;
  v4u g0;
  g0[0] = 1u;                                   // count=1 (valid user descriptor)
  g0[1] = lds_off;                              // lds_addr
  g0[2] = (unsigned)ga;                         // global_addr[31:0]
  g0[3] = (unsigned)(ga >> 32) | 0x80000000u;   // global_addr[56:32] | type=2
  v8i g1;
  g1[0] = 0x00010000;   // workgroup_mask=0, data_size=1 (2 bytes)
  g1[1] = 0x20000000;   // tensor_dim0 = 8192 (low 16 bits at [63:48])
  g1[2] = 0x00010000;   // tensor_dim0 hi = 0, tensor_dim1 = 1
  g1[3] = 0x20000000;   // tensor_dim1 hi = 0, tile_dim0 = 8192
  g1[4] = 1;            // tile_dim1 = 1, tile_dim2 = 0
  g1[5] = 8192;         // tensor_dim0_stride low32
  g1[6] = 0;
  g1[7] = 0;
  v4i z4 = {0, 0, 0, 0};
#if __clang_major__ >= 23
  v8i z8 = {0, 0, 0, 0, 0, 0, 0, 0};
  __builtin_amdgcn_tensor_load_to_lds(g0, g1, z4, z4, z8, 0);
#else
  __builtin_amdgcn_tensor_load_to_lds(g0, g1, z4, z4, 0);
#endif
}

// ---------------- WMMA fragment loaders (contiguous k = 16*(lane/16) + j in A and B) ----------------

// A matrix 16x32 bf16, source row-major [m][k] with given pitch (elements).
__device__ inline v16bf load_a_frag(const bf16_t* src, int pitch) {
  int lane = threadIdx.x & 31;
  int m = lane & 15;
  int kh = (lane >> 4) & 1;
  const bf16_t* row = src + m * pitch + kh * 16;
  v16bf a;
#pragma unroll
  for (int j = 0; j < 16; ++j) {
    int k = j;
    a[j] = row[k];
  }
  return a;
}

// Same but source is f32 (softmax probabilities), converted to bf16 on load.
__device__ inline v16bf load_a_frag_f32(const float* src, int pitch) {
  int lane = threadIdx.x & 31;
  int m = lane & 15;
  int kh = (lane >> 4) & 1;
  const float* row = src + m * pitch + kh * 16;
  v16bf a;
#pragma unroll
  for (int j = 0; j < 16; ++j) {
    int k = j;
    a[j] = (bf16_t)row[k];
  }
  return a;
}

// B matrix 32x16 bf16, B[k][n], source stored [n][k].
__device__ inline v16bf load_b_frag_nk(const bf16_t* src, int pitch) {
  int lane = threadIdx.x & 31;
  int n = lane & 15;
  int k0 = (lane >> 4) ? 16 : 0;
  const bf16_t* col = src + n * pitch + k0;
  v16bf b;
#pragma unroll
  for (int j = 0; j < 16; ++j) b[j] = col[j];
  return b;
}

// B matrix 32x16 bf16, source stored [k][n].
__device__ inline v16bf load_b_frag_kn(const bf16_t* src, int pitch) {
  int lane = threadIdx.x & 31;
  int n = lane & 15;
  int k0 = (lane >> 4) ? 16 : 0;
  v16bf b;
#pragma unroll
  for (int j = 0; j < 16; ++j) b[j] = src[(k0 + j) * pitch + n];
  return b;
}

__device__ inline v8f wmma_bf16(v16bf a, v16bf b, v8f c) {
  return __builtin_amdgcn_wmma_f32_16x16x32_bf16(false, a, false, b, (short)0, c,
                                                 false, false);
}

// ---------------- Kernel 1: LayerNorm -> bf16 ----------------
__global__ __launch_bounds__(256) void ln_kernel(const float* __restrict__ m,
                                                 const float* __restrict__ lnw,
                                                 const float* __restrict__ lnb,
                                                 bf16_t* __restrict__ xn) {
  int w = threadIdx.x >> 5, lane = threadIdx.x & 31;
  int row = blockIdx.x * 8 + w;
  const float* src = m + (size_t)row * C_DIM + lane * 8;
  float4 x0 = ((const float4*)src)[0];
  float4 x1 = ((const float4*)src)[1];
  float xs[8] = {x0.x, x0.y, x0.z, x0.w, x1.x, x1.y, x1.z, x1.w};
  float s = 0.f, q = 0.f;
#pragma unroll
  for (int j = 0; j < 8; ++j) { s += xs[j]; q += xs[j] * xs[j]; }
#pragma unroll
  for (int off = 16; off > 0; off >>= 1) {
    s += __shfl_xor(s, off, 32);
    q += __shfl_xor(q, off, 32);
  }
  float mean = s * (1.f / C_DIM);
  float var = q * (1.f / C_DIM) - mean * mean;
  float rs = rsqrtf(var + 1e-5f);
  float4 w0 = ((const float4*)(lnw + lane * 8))[0];
  float4 w1 = ((const float4*)(lnw + lane * 8))[1];
  float4 b0 = ((const float4*)(lnb + lane * 8))[0];
  float4 b1 = ((const float4*)(lnb + lane * 8))[1];
  float lw[8] = {w0.x, w0.y, w0.z, w0.w, w1.x, w1.y, w1.z, w1.w};
  float lb[8] = {b0.x, b0.y, b0.z, b0.w, b1.x, b1.y, b1.z, b1.w};
  union { uint4 u; bf16_t h[8]; } pk;
#pragma unroll
  for (int j = 0; j < 8; ++j)
    pk.h[j] = (bf16_t)((xs[j] - mean) * rs * lw[j] + lb[j]);
  *(uint4*)(xn + (size_t)row * C_DIM + lane * 8) = pk.u;
}

// ---------------- Kernel 2: fused Q/K/V/G projection GEMM ----------------
#define APITCH 40
#define BPITCH 40
__global__ __launch_bounds__(256) void qkvg_kernel(
    const bf16_t* __restrict__ xn, const float* __restrict__ wq,
    const float* __restrict__ wk, const float* __restrict__ wv,
    const float* __restrict__ wg, const float* __restrict__ bg,
    const float* __restrict__ mask, bf16_t* __restrict__ qws,
    bf16_t* __restrict__ kws, bf16_t* __restrict__ vws,
    bf16_t* __restrict__ gws) {
  __shared__ bf16_t As[64 * APITCH];
  __shared__ bf16_t Bs[4][32 * BPITCH];
  int tid = threadIdx.x;
  int w = tid >> 5, mw = w & 3, nw = w >> 2;
  int row0 = blockIdx.x * 64, col0 = blockIdx.y * 32;
  v8f acc0 = {}, acc1 = {}, acc2 = {}, acc3 = {};

  // loop-invariant staging indices / pointers (hoisted out of the K loop)
  int ar = tid >> 2, akk = (tid & 3) * 8;
  unsigned a_lds = to_lds(As + ar * APITCH + akk);
  const bf16_t* a_gp = xn + (size_t)(row0 + ar) * C_DIM + akk;
  int wsel = tid >> 6, tt = tid & 63, bo_ = tt >> 1, bkk = (tt & 1) * 16;
  const float* srcw = (wsel == 0) ? wq : (wsel == 1) ? wk : (wsel == 2) ? wv : wg;
  const float* b_gp = srcw + (size_t)(col0 + bo_) * C_DIM + bkk;

  for (int ks = 0; ks < 8; ++ks) {
    // A tile [64 x 32] bf16: async global -> LDS, no VGPR staging
    async_copy_b128(a_lds, a_gp + ks * 32);
    if (ks < 7) __builtin_prefetch(a_gp + (ks + 1) * 32, 0, 1);
    // 4 weight tiles [32 o x 32 k] f32 -> bf16
    {
      const float* src = b_gp + ks * 32;
#pragma unroll
      for (int j = 0; j < 16; ++j) Bs[wsel][bo_ * BPITCH + bkk + j] = (bf16_t)src[j];
    }
    async_wait0();
    __syncthreads();
    v16bf a = load_a_frag(As + mw * 16 * APITCH, APITCH);
    {
      v16bf b = load_b_frag_nk(&Bs[0][nw * 16 * BPITCH], BPITCH);
      acc0 = wmma_bf16(a, b, acc0);
    }
    {
      v16bf b = load_b_frag_nk(&Bs[1][nw * 16 * BPITCH], BPITCH);
      acc1 = wmma_bf16(a, b, acc1);
    }
    {
      v16bf b = load_b_frag_nk(&Bs[2][nw * 16 * BPITCH], BPITCH);
      acc2 = wmma_bf16(a, b, acc2);
    }
    {
      v16bf b = load_b_frag_nk(&Bs[3][nw * 16 * BPITCH], BPITCH);
      acc3 = wmma_bf16(a, b, acc3);
    }
    __syncthreads();
  }
  // epilogue: mask, scale into Q, sigmoid gate, scatter to [i,h,s,d] bf16
  int lane = tid & 31;
  int n = lane & 15;
  int mb = (lane >> 4) ? 8 : 0;
#pragma unroll
  for (int r = 0; r < 8; ++r) {
    int grow = row0 + mw * 16 + mb + r;       // = s*256 + i
    int gcol = col0 + nw * 16 + n;            // = h*32 + dd
    int s_ = grow >> 8, i_ = grow & 255;
    int h_ = gcol >> 5, dd = gcol & 31;
    float mk = mask[grow] > 0.f ? 1.f : 0.f;
    size_t di = ((size_t)(i_ * H_DIM + h_) * S_DIM + s_) * D_DIM + dd;
    qws[di] = (bf16_t)(acc0[r] * mk * 0.17677669529663687f);  // 1/sqrt(32)
    kws[di] = (bf16_t)(acc1[r] * mk);
    vws[di] = (bf16_t)(acc2[r] * mk);
    float gg = acc3[r] + bg[gcol];
    gws[di] = (bf16_t)(mk / (1.f + __expf(-gg)));
  }
}

// ---------------- Kernel 3: attention per (i, h), whole problem in LDS ----------------
__global__ __launch_bounds__(256) void attn_kernel(
    const bf16_t* __restrict__ qws, const bf16_t* __restrict__ kws,
    const bf16_t* __restrict__ vws, const bf16_t* __restrict__ gws,
    const float* __restrict__ mask, bf16_t* __restrict__ ows) {
  extern __shared__ char smem[];
  bf16_t* Qs = (bf16_t*)smem;                      // 256 x 32 bf16
  bf16_t* Ks = Qs + S_DIM * D_DIM;
  bf16_t* Vs = Ks + S_DIM * D_DIM;
  float* Sb = (float*)(Vs + S_DIM * D_DIM);        // 256 x SP f32
  float* kb = Sb + S_DIM * SP;                     // 256 f32 key bias

  int h_ = blockIdx.x, i_ = blockIdx.y;
  int tid = threadIdx.x, w = tid >> 5, lane = tid & 31;
  size_t base = (size_t)(i_ * H_DIM + h_) * S_DIM * D_DIM;

  // Q/K/V slabs (16KB each, contiguous) staged by the Tensor Data Mover.
  if (w == 0) {
    tdm_load_16kb(to_lds(Qs), qws + base);
    tdm_load_16kb(to_lds(Ks), kws + base);
    tdm_load_16kb(to_lds(Vs), vws + base);
    __builtin_amdgcn_s_wait_tensorcnt(0);
  }
  kb[tid] = mask[tid * I_DIM + i_] > 0.f ? 0.f : -1e9f;
  __syncthreads();

  // S = Q K^T (scale folded into Q) + key bias
  int m0 = w * 32;
  int n = lane & 15;
  int mb = (lane >> 4) ? 8 : 0;
#pragma unroll
  for (int mt = 0; mt < 2; ++mt) {
    v16bf a = load_a_frag(Qs + (m0 + mt * 16) * D_DIM, D_DIM);
    for (int nt = 0; nt < 16; ++nt) {
      v8f c = {};
      v16bf b = load_b_frag_nk(Ks + nt * 16 * D_DIM, D_DIM);
      c = wmma_bf16(a, b, c);
      float bias = kb[nt * 16 + n];
#pragma unroll
      for (int r = 0; r < 8; ++r)
        Sb[(m0 + mt * 16 + mb + r) * SP + nt * 16 + n] = c[r] + bias;
    }
  }
  __syncthreads();

  // softmax: one thread per query row
  {
    float* rowp = Sb + tid * SP;
    float mx = -3.4e38f;
    for (int k = 0; k < S_DIM; ++k) mx = fmaxf(mx, rowp[k]);
    float sum = 0.f;
    for (int k = 0; k < S_DIM; ++k) {
      float e = __expf(rowp[k] - mx);
      rowp[k] = e;
      sum += e;
    }
    float inv = 1.f / sum;
    for (int k = 0; k < S_DIM; ++k) rowp[k] *= inv;
  }
  __syncthreads();

  // O = P V, gate, store to [i, s, h*32+d] bf16
#pragma unroll
  for (int mt = 0; mt < 2; ++mt) {
    v8f acc0 = {}, acc1 = {};
    for (int ks = 0; ks < 8; ++ks) {
      v16bf a = load_a_frag_f32(Sb + (m0 + mt * 16) * SP + ks * 32, SP);
      v16bf b0 = load_b_frag_kn(Vs + ks * 32 * D_DIM, D_DIM);
      v16bf b1 = load_b_frag_kn(Vs + ks * 32 * D_DIM + 16, D_DIM);
      acc0 = wmma_bf16(a, b0, acc0);
      acc1 = wmma_bf16(a, b1, acc1);
    }
#pragma unroll
    for (int r = 0; r < 8; ++r) {
      int srow = m0 + mt * 16 + mb + r;
      size_t orow = ((size_t)(i_ * S_DIM) + srow) * C_DIM + h_ * D_DIM;
      {
        size_t gi = base + (size_t)srow * D_DIM + n;
        ows[orow + n] = (bf16_t)(acc0[r] * (float)gws[gi]);
      }
      {
        size_t gi = base + (size_t)srow * D_DIM + 16 + n;
        ows[orow + 16 + n] = (bf16_t)(acc1[r] * (float)gws[gi]);
      }
    }
  }
}

// ---------------- Kernel 4: output projection + bias, transposed store ----------------
__global__ __launch_bounds__(256) void oproj_kernel(const bf16_t* __restrict__ ows,
                                                    const float* __restrict__ wo,
                                                    const float* __restrict__ bo,
                                                    float* __restrict__ out) {
  __shared__ bf16_t As[64 * APITCH];
  __shared__ bf16_t Bs[32 * BPITCH];
  int tid = threadIdx.x;
  int w = tid >> 5, mw = w & 3, nw = w >> 2;
  int row0 = blockIdx.x * 64, col0 = blockIdx.y * 32;
  v8f acc = {};

  int ar = tid >> 2, akk = (tid & 3) * 8;
  unsigned a_lds = to_lds(As + ar * APITCH + akk);
  const bf16_t* a_gp = ows + (size_t)(row0 + ar) * C_DIM + akk;
  int bo2 = tid >> 3, bkk = (tid & 7) * 4;
  const float* b_gp = wo + (size_t)(col0 + bo2) * C_DIM + bkk;

  for (int ks = 0; ks < 8; ++ks) {
    async_copy_b128(a_lds, a_gp + ks * 32);
    {
      const float* src = b_gp + ks * 32;
#pragma unroll
      for (int j = 0; j < 4; ++j) Bs[bo2 * BPITCH + bkk + j] = (bf16_t)src[j];
    }
    async_wait0();
    __syncthreads();
    v16bf a = load_a_frag(As + mw * 16 * APITCH, APITCH);
    v16bf b = load_b_frag_nk(Bs + nw * 16 * BPITCH, BPITCH);
    acc = wmma_bf16(a, b, acc);
    __syncthreads();
  }
  int lane = tid & 31;
  int n = lane & 15;
  int mb = (lane >> 4) ? 8 : 0;
#pragma unroll
  for (int r = 0; r < 8; ++r) {
    int grow = row0 + mw * 16 + mb + r;  // = i*256 + s
    int gcol = col0 + nw * 16 + n;
    int i_ = grow >> 8, s_ = grow & 255;
    out[((size_t)s_ * I_DIM + i_) * C_DIM + gcol] = acc[r] + bo[gcol];
  }
}

// ---------------- launch ----------------
extern "C" void kernel_launch(void* const* d_in, const int* in_sizes, int n_in,
                              void* d_out, int out_size, void* d_ws, size_t ws_size,
                              hipStream_t stream) {
  const float* m    = (const float*)d_in[0];
  const float* mask = (const float*)d_in[1];
  const float* lnw  = (const float*)d_in[2];
  const float* lnb  = (const float*)d_in[3];
  const float* wq   = (const float*)d_in[4];
  const float* wk   = (const float*)d_in[5];
  const float* wv   = (const float*)d_in[6];
  const float* wg   = (const float*)d_in[7];
  const float* bg   = (const float*)d_in[8];
  const float* wo   = (const float*)d_in[9];
  const float* bo   = (const float*)d_in[10];
  float* out = (float*)d_out;

  const size_t M = (size_t)NROW * C_DIM;  // 16.7M elems
  bf16_t* xn  = (bf16_t*)d_ws;
  bf16_t* qws = xn + M;
  bf16_t* kws = qws + M;
  bf16_t* vws = kws + M;
  bf16_t* gws = vws + M;
  bf16_t* ows = gws + M;

  ln_kernel<<<NROW / 8, 256, 0, stream>>>(m, lnw, lnb, xn);

  qkvg_kernel<<<dim3(NROW / 64, C_DIM / 32), 256, 0, stream>>>(
      xn, wq, wk, wv, wg, bg, mask, qws, kws, vws, gws);

  size_t attn_lds = (size_t)3 * S_DIM * D_DIM * sizeof(bf16_t) +
                    (size_t)S_DIM * SP * sizeof(float) + 256 * sizeof(float);
  hipFuncSetAttribute((const void*)attn_kernel,
                      hipFuncAttributeMaxDynamicSharedMemorySize, (int)attn_lds);
  attn_kernel<<<dim3(H_DIM, I_DIM), 256, attn_lds, stream>>>(qws, kws, vws, gws,
                                                             mask, ows);

  oproj_kernel<<<dim3(NROW / 64, C_DIM / 32), 256, 0, stream>>>(ows, wo, bo, out);
}
